// PrototypeKmeansDivergence_50560355009195
// MI455X (gfx1250) — matrix operand order repaired and checked by reading, not compiled, not run on hardware
//
#include <hip/hip_runtime.h>
#include <hip/hip_bf16.h>

// ---------------------------------------------------------------------------
// Problem constants (match reference)
// ---------------------------------------------------------------------------
static constexpr int BATCH = 32;
static constexpr int T     = 1024;
static constexpr int D     = 1024;
static constexpr int KST   = 10;    // STATE_NUM == K
static constexpr int KPAD  = 16;    // padded centers for 16-wide WMMA tiles
static constexpr int ITERS = 11;    // 10 scan steps + final recompute

typedef __attribute__((ext_vector_type(16))) __bf16 v16bf;
typedef __attribute__((ext_vector_type(8)))  float  v8f;

// ---------------------------------------------------------------------------
// Workspace layout (bytes)
// ---------------------------------------------------------------------------
static constexpr size_t OFF_FEATB   = 0;                                        // bf16 features  B*T*D*2 = 64MB
static constexpr size_t SZ_FEATB    = (size_t)BATCH * T * D * 2;
static constexpr size_t OFF_CENT    = OFF_FEATB + SZ_FEATB;                     // f32 centers B*16*D*4
static constexpr size_t SZ_CENT     = (size_t)BATCH * KPAD * D * 4;
static constexpr size_t OFF_CENTB   = OFF_CENT + SZ_CENT;                       // bf16 centers
static constexpr size_t SZ_CENTB    = (size_t)BATCH * KPAD * D * 2;
static constexpr size_t OFF_CC      = OFF_CENTB + SZ_CENTB;                     // ||c||^2  B*16
static constexpr size_t SZ_CC       = (size_t)BATCH * KPAD * 4;
static constexpr size_t OFF_PROTO   = OFF_CC + 4096;                            // f32 protos B*10*D
static constexpr size_t SZ_PROTO    = (size_t)BATCH * KST * D * 4;
static constexpr size_t OFF_ASG     = OFF_PROTO + SZ_PROTO;                     // int assign B*T
static constexpr size_t SZ_ASG      = (size_t)BATCH * T * 4;
static constexpr size_t OFF_PD      = OFF_ASG + SZ_ASG;                         // per-batch loss B

// ---------------------------------------------------------------------------
// K0: convert f32 features -> bf16 (8 elements / thread, vectorized)
// ---------------------------------------------------------------------------
__global__ __launch_bounds__(256)
void pkd_cvt_bf16(const float* __restrict__ x, __bf16* __restrict__ y, long n)
{
    long i = ((long)blockIdx.x * blockDim.x + threadIdx.x) * 8;
    if (i + 8 > n) return;
    float4 f0 = *reinterpret_cast<const float4*>(x + i);
    float4 f1 = *reinterpret_cast<const float4*>(x + i + 4);
    union { __bf16 h[8]; uint4 q; } o;
    o.h[0] = (__bf16)f0.x; o.h[1] = (__bf16)f0.y; o.h[2] = (__bf16)f0.z; o.h[3] = (__bf16)f0.w;
    o.h[4] = (__bf16)f1.x; o.h[5] = (__bf16)f1.y; o.h[6] = (__bf16)f1.z; o.h[7] = (__bf16)f1.w;
    *reinterpret_cast<uint4*>(y + i) = o.q;
}

// ---------------------------------------------------------------------------
// K1: per-batch label prototypes: protos[s][d] = sum_{t: lab==s} x[t][d] / max(cnt,1)
// one block per batch, 1024 threads (thread = feature dim d)
// ---------------------------------------------------------------------------
__global__ __launch_bounds__(1024)
void pkd_protos(const float* __restrict__ x, const int* __restrict__ labels,
                float* __restrict__ protos)
{
    __shared__ int slab[T];
    __shared__ int scnt[KPAD];
    const int b = blockIdx.x, tid = threadIdx.x;
    if (tid < KPAD) scnt[tid] = 0;
    slab[tid] = labels[(size_t)b * T + tid];
    __syncthreads();
    atomicAdd(&scnt[slab[tid]], 1);

    float acc[KST];
#pragma unroll
    for (int s = 0; s < KST; ++s) acc[s] = 0.f;

    const float* xb = x + (size_t)b * T * D + tid;   // column tid
    for (int t = 0; t < T; ++t) {
        int l = slab[t];
        float v = xb[(size_t)t * D];
#pragma unroll
        for (int s = 0; s < KST; ++s) acc[s] += (l == s) ? v : 0.f;
    }
    __syncthreads();
#pragma unroll
    for (int s = 0; s < KST; ++s)
        protos[((size_t)b * KST + s) * D + tid] = acc[s] / fmaxf((float)scnt[s], 1.f);
}

// ---------------------------------------------------------------------------
// K2: initialize centers = x[:10] (rows 10..15 zero), bf16 copy, and ||c||^2
// ---------------------------------------------------------------------------
__global__ __launch_bounds__(256)
void pkd_init_centers(const float* __restrict__ x, float* __restrict__ cent,
                      __bf16* __restrict__ centB, float* __restrict__ cc)
{
    __shared__ float sq[KPAD];
    const int b = blockIdx.x, tid = threadIdx.x;
    if (tid < KPAD) sq[tid] = 0.f;
    __syncthreads();
    for (int idx = tid; idx < KPAD * D; idx += 256) {
        int n = idx >> 10, d = idx & (D - 1);
        float v = (n < KST) ? x[((size_t)b * T + n) * D + d] : 0.f;
        cent [((size_t)b * KPAD + n) * D + d] = v;
        centB[((size_t)b * KPAD + n) * D + d] = (__bf16)v;
        atomicAdd(&sq[n], v * v);
    }
    __syncthreads();
    if (tid < KPAD) cc[(size_t)b * KPAD + tid] = sq[tid];
}

// ---------------------------------------------------------------------------
// K3: WMMA assignment. score(m,n) = ||c_n||^2 - 2 * x_m . c_n ; argmin over n<10.
// block = 128 threads = 4 waves; each wave owns a 16-row tile; grid (B, 16).
// Centers live in LDS as scT[n][k] (centB order) so the WMMA B-matrix (32x16 bf16,
// lane = N column, two runs of 8 contiguous K per lane) is two ds_load_b128 per k-step.
// ---------------------------------------------------------------------------
__global__ __launch_bounds__(128)
void pkd_assign(const __bf16* __restrict__ featB, const __bf16* __restrict__ centB,
                const float* __restrict__ cc, int* __restrict__ asg)
{
    __shared__ __bf16 scT[D * KPAD];   // 32 KB, [n][k]
    __shared__ float  scc[KPAD];
    const int b = blockIdx.x, tid = threadIdx.x;

    // cooperative copy of centers into LDS (coalesced global reads)
    for (int idx = tid; idx < KPAD * D; idx += 128) {
        int n = idx >> 10, k = idx & (D - 1);
        scT[n * D + k] = centB[((size_t)b * KPAD + n) * D + k];
    }
    if (tid < KPAD) scc[tid] = cc[(size_t)b * KPAD + tid];
    __syncthreads();

    const int wave = tid >> 5;
    const int lane = tid & 31;
    const int half = lane >> 4;        // K-half selector for A, M-half for C/D
    const int nidx = lane & 15;        // A row within tile / C column
    const int m0   = (blockIdx.y * 4 + wave) * 16;

    const __bf16* arow = featB + ((size_t)b * T + (m0 + nidx)) * D;

    v8f acc = {};
    for (int k0 = 0; k0 < D; k0 += 32) {
        // A (16x32 bf16): lane holds K = 8h..8h+7 and 16+8h..23+8h of its row
        union { v16bf v; uint4 q[2]; } A, Bm;
        A.q[0] = *reinterpret_cast<const uint4*>(arow + k0 + half * 8);
        A.q[1] = *reinterpret_cast<const uint4*>(arow + k0 + 16 + half * 8);
        // B (32x16 bf16): lane holds column N=nidx, K = 8h..8h+7 and 16+8h..23+8h
        const __bf16* brow = scT + nidx * D + k0;
        Bm.q[0] = *reinterpret_cast<const uint4*>(brow + half * 8);
        Bm.q[1] = *reinterpret_cast<const uint4*>(brow + 16 + half * 8);
        acc = __builtin_amdgcn_wmma_f32_16x16x32_bf16(
                  false, A.v, false, Bm.v, (short)0, acc, false, false);
    }

    // C/D layout: VGPR v -> M = v + 8*half, N = nidx. Argmin across N within halves.
    const float myc = scc[nidx];
    int best[8];
#pragma unroll
    for (int v = 0; v < 8; ++v) {
        float s = (nidx < KST) ? (myc - 2.0f * acc[v]) : 3.0e38f;
        int   bi = nidx;
#pragma unroll
        for (int off = 8; off >= 1; off >>= 1) {          // stays inside 16-lane half
            float os = __shfl_xor(s, off, 32);
            int   oi = __shfl_xor(bi, off, 32);
            if (os < s || (os == s && oi < bi)) { s = os; bi = oi; }
        }
        best[v] = bi;
    }
    if (nidx == 0) {
#pragma unroll
        for (int v = 0; v < 8; ++v)
            asg[(size_t)b * T + m0 + half * 8 + v] = best[v];
    }
}

// ---------------------------------------------------------------------------
// K4: centroid update: newc = (cnt>0) ? sums/cnt : oldc ; refresh bf16 + ||c||^2
// one block per batch, 1024 threads (thread = dim d)
// ---------------------------------------------------------------------------
__global__ __launch_bounds__(1024)
void pkd_update(const float* __restrict__ x, const int* __restrict__ asg,
                float* __restrict__ cent, __bf16* __restrict__ centB,
                float* __restrict__ cc)
{
    __shared__ int   sasg[T];
    __shared__ int   scnt[KPAD];
    __shared__ float sq[KPAD];
    const int b = blockIdx.x, tid = threadIdx.x;
    if (tid < KPAD) { scnt[tid] = 0; sq[tid] = 0.f; }
    sasg[tid] = asg[(size_t)b * T + tid];
    __syncthreads();
    atomicAdd(&scnt[sasg[tid]], 1);

    float acc[KST];
#pragma unroll
    for (int s = 0; s < KST; ++s) acc[s] = 0.f;

    const float* xb = x + (size_t)b * T * D + tid;
    for (int t = 0; t < T; ++t) {
        int a = sasg[t];
        float v = xb[(size_t)t * D];
#pragma unroll
        for (int s = 0; s < KST; ++s) acc[s] += (a == s) ? v : 0.f;
    }
    __syncthreads();
#pragma unroll
    for (int s = 0; s < KST; ++s) {
        size_t o = ((size_t)b * KPAD + s) * D + tid;
        int c = scnt[s];
        float nc = (c > 0) ? (acc[s] / (float)c) : cent[o];
        cent[o]  = nc;
        centB[o] = (__bf16)nc;
        atomicAdd(&sq[s], nc * nc);
    }
    __syncthreads();
    if (tid < KPAD) cc[(size_t)b * KPAD + tid] = sq[tid];
}

// ---------------------------------------------------------------------------
// K5: per-batch epilogue: 10x10 distances, top-3 map, iterative Kuhn matching,
// matched-distance mean -> per_dlg[b]
// ---------------------------------------------------------------------------
__global__ __launch_bounds__(128)
void pkd_final(const float* __restrict__ protos, const float* __restrict__ cent,
               float* __restrict__ perdlg)
{
    __shared__ float dl[KST][KST];
    __shared__ int   mp[KST][KST];
    const int b = blockIdx.x, tid = threadIdx.x;

    if (tid < KST * KST) {
        int s = tid / KST, k = tid % KST;
        const float* pr = protos + ((size_t)b * KST + s) * D;
        const float* ce = cent   + ((size_t)b * KPAD + k) * D;
        float sum = 0.f;
        for (int d = 0; d < D; ++d) { float df = pr[d] - ce[d]; sum += df * df; }
        dl[s][k] = sqrtf(fmaxf(sum, 0.f));
    }
    __syncthreads();

    if (tid < KST) {                                  // top-3 largest per row
        int s = tid;
        bool used[KST];
#pragma unroll
        for (int k = 0; k < KST; ++k) { used[k] = false; mp[s][k] = 0; }
        for (int r = 0; r < 3; ++r) {
            float best = -3.0e38f; int bi = 0;
            for (int k = 0; k < KST; ++k)
                if (!used[k] && dl[s][k] > best) { best = dl[s][k]; bi = k; }
            used[bi] = true; mp[s][bi] = 1;
        }
    }
    __syncthreads();

    if (tid == 0) {                                   // iterative Kuhn's matching
        int p[KST]; int cnt = 0;
#pragma unroll
        for (int k = 0; k < KST; ++k) p[k] = 0;
        for (int i0 = 0; i0 < KST; ++i0) {
            bool vis[KST];
#pragma unroll
            for (int k = 0; k < KST; ++k) vis[k] = false;
            int rowS[KST + 1], jS[KST + 1];
            int sp = 0; rowS[0] = i0; jS[0] = 0; bool ok = false;
            for (;;) {
                int r = rowS[sp], j = jS[sp], found = -1;
                for (; j < KST; ++j)
                    if (mp[r][j] && !vis[j]) { found = j; break; }
                if (found < 0) {
                    if (sp == 0) { ok = false; break; }
                    --sp; jS[sp]++;                   // parent resumes after its column
                    continue;
                }
                vis[found] = true;
                if (p[found] == 0) {
                    p[found] = r + 1;
                    while (sp > 0) { --sp; p[jS[sp]] = rowS[sp] + 1; }
                    ok = true; break;
                }
                jS[sp] = found;
                ++sp; rowS[sp] = p[found] - 1; jS[sp] = 0;
            }
            if (ok) ++cnt;
        }
        float acc = 0.f;
        for (int k = 0; k < KST; ++k)
            if (p[k] > 0) acc += dl[p[k] - 1][k];
        int cm = (cnt > 1) ? cnt : 1;
        perdlg[b] = acc / (float)cm;
    }
}

// K6: mean over batches -> scalar out
__global__ __launch_bounds__(32)
void pkd_mean(const float* __restrict__ perdlg, float* __restrict__ out)
{
    float v = perdlg[threadIdx.x];                    // BATCH == 32 == wave32
#pragma unroll
    for (int off = 16; off >= 1; off >>= 1) v += __shfl_xor(v, off, 32);
    if (threadIdx.x == 0) out[0] = v / (float)BATCH;
}

// ---------------------------------------------------------------------------
extern "C" void kernel_launch(void* const* d_in, const int* in_sizes, int n_in,
                              void* d_out, int out_size, void* d_ws, size_t ws_size,
                              hipStream_t stream)
{
    const float* features = (const float*)d_in[0];
    // d_in[1] = dialogue_lengths (all == T; unused, as in reference)
    const int*   labels   = (const int*)d_in[2];

    char* w = (char*)d_ws;
    __bf16* featB  = (__bf16*)(w + OFF_FEATB);
    float*  cent   = (float*) (w + OFF_CENT);
    __bf16* centB  = (__bf16*)(w + OFF_CENTB);
    float*  cc     = (float*) (w + OFF_CC);
    float*  protos = (float*) (w + OFF_PROTO);
    int*    asg    = (int*)   (w + OFF_ASG);
    float*  perdlg = (float*) (w + OFF_PD);

    const long nfeat = (long)BATCH * T * D;
    pkd_cvt_bf16<<<(int)(nfeat / (256 * 8)), 256, 0, stream>>>(features, featB, nfeat);
    pkd_protos<<<BATCH, 1024, 0, stream>>>(features, labels, protos);
    pkd_init_centers<<<BATCH, 256, 0, stream>>>(features, cent, centB, cc);

    for (int it = 0; it < ITERS; ++it) {
        pkd_assign<<<dim3(BATCH, T / 64), 128, 0, stream>>>(featB, centB, cc, asg);
        pkd_update<<<BATCH, 1024, 0, stream>>>(features, asg, cent, centB, cc);
    }

    pkd_final<<<BATCH, 128, 0, stream>>>(protos, cent, perdlg);
    pkd_mean<<<1, 32, 0, stream>>>(perdlg, (float*)d_out);
}
